// ConditionalGRAN_18940805775701
// MI455X (gfx1250) — compile-verified, matrix operand order checked by reading, not run on hardware
//
#include <hip/hip_runtime.h>
#include <hip/hip_bf16.h>
#include <math.h>

// Problem constants (match reference)
#define NODE_DIM 128
#define LAT      64
#define KCOMB    192           // NODE_DIM + LAT
#define HID      256
#define HEADS    4
#define CPH      64            // channels per head
#define NEG_SLOPE 0.2f

typedef _Float16 v16h __attribute__((ext_vector_type(16)));
typedef _Float16 half8 __attribute__((ext_vector_type(8)));
typedef float    v8f  __attribute__((ext_vector_type(8)));

// ---------------------------------------------------------------------------
// WMMA fragment index helpers (CDNA5 ISA 7.12.2, wave32)
// A (16x32 f16): lanes 0-15 -> M=lane, halves 0-7 = K 0..7, halves 8-15 = K 16..23
//                lanes 16-31 -> M=lane-16, halves 0-7 = K 8..15, 8-15 = K 24..31
// D (16x16 f32): elem r -> M = (lane<16 ? r : r+8), N = lane & 15
// B is stored pre-swizzled by the producer kernels with the mirrored mapping:
//   lane L, half j -> n = L&15, k = (L<16) ? (j<8?j:j+8) : (j<8?j+8:j+16)
// ---------------------------------------------------------------------------

__device__ __forceinline__ int b_koff(int lane, int j) {
    return (lane < 16) ? ((j < 8) ? j : j + 8) : ((j < 8) ? j + 8 : j + 16);
}

__device__ __forceinline__ void atomicMaxF(float* addr, float v) {
    if (v >= 0.0f) atomicMax((int*)addr, __float_as_int(v));
    else           atomicMin((unsigned int*)addr, __float_as_uint(v));
}

// ---------------------------------------------------------------------------
// CDNA5 async global->LDS copy (ASYNCcnt-tracked, no VGPR round trip).
// GV mode: per-lane 64-bit global address, per-lane 32-bit LDS byte address.
// Low 32 bits of a generic pointer to LDS are the LDS byte offset (ISA 10.2).
// ---------------------------------------------------------------------------
__device__ __forceinline__ void async_copy_b128(void* lds_ptr, const void* gptr) {
    unsigned lds_off = (unsigned)(unsigned long long)(uintptr_t)lds_ptr;
    unsigned long long ga = (unsigned long long)(uintptr_t)gptr;
    asm volatile("global_load_async_to_lds_b128 %0, %1, off"
                 :: "v"(lds_off), "v"(ga) : "memory");
}
__device__ __forceinline__ void wait_async0() {
    asm volatile("s_wait_asynccnt 0" ::: "memory");
}

// Build an A fragment from an f32 LDS panel (row-major, row_stride floats).
__device__ __forceinline__ v16h make_afrag(const float* Ap, int row_stride,
                                           int am, int s, int klo) {
    const float* p = Ap + am * row_stride + s * 32 + klo;
    v16h a;
    #pragma unroll
    for (int i = 0; i < 8; ++i) a[i]     = (_Float16)p[i];
    #pragma unroll
    for (int i = 0; i < 8; ++i) a[i + 8] = (_Float16)p[i + 16];
    return a;
}

// ---------------------------------------------------------------------------
// K0a: W_comb_sw[f16, B-fragment order] = (W_gat @ [W_enc | W_z]) swizzled.
// 16 N-tiles x 6 K-steps x 32 lanes x 16 halves = 49152 elements.
// ---------------------------------------------------------------------------
__global__ void prep_wcomb(const float* __restrict__ W_gat,
                           const float* __restrict__ W_enc,
                           const float* __restrict__ W_z,
                           _Float16* __restrict__ wsw) {
    int idx = blockIdx.x * blockDim.x + threadIdx.x;
    if (idx >= 16 * 6 * 32 * 16) return;
    int j    = idx & 15;
    int lane = (idx >> 4) & 31;
    int s    = (idx >> 9) % 6;
    int t    = (idx >> 9) / 6;
    int o = t * 16 + (lane & 15);       // output column (N)
    int k = s * 32 + b_koff(lane, j);   // input row (K), 0..191
    float acc = 0.0f;
    if (k < NODE_DIM) {
        for (int q = 0; q < HID; ++q)
            acc += W_gat[o * HID + q] * W_enc[q * NODE_DIM + k];
    } else {
        int kz = k - NODE_DIM;
        for (int q = 0; q < HID; ++q)
            acc += W_gat[o * HID + q] * W_z[q * LAT + kz];
    }
    wsw[idx] = (_Float16)acc;
}

// K0b: W_e1 swizzled to B-fragment order. 16 tiles x 8 steps x 32 x 16 = 65536.
__global__ void prep_we1(const float* __restrict__ W_e1, _Float16* __restrict__ wsw) {
    int idx = blockIdx.x * blockDim.x + threadIdx.x;
    if (idx >= 16 * 8 * 32 * 16) return;
    int j    = idx & 15;
    int lane = (idx >> 4) & 31;
    int s    = (idx >> 9) & 7;
    int t    = (idx >> 9) >> 3;
    int o = t * 16 + (lane & 15);
    int k = s * 32 + b_koff(lane, j);
    wsw[idx] = (_Float16)W_e1[o * HID + k];
}

// K0c: bvec = W_gat @ (b_enc + b_z)   (bias folded through the GAT linear)
__global__ void prep_bvec(const float* __restrict__ W_gat,
                          const float* __restrict__ b_enc,
                          const float* __restrict__ b_z,
                          float* __restrict__ bvec) {
    int o = threadIdx.x;
    float acc = 0.0f;
    for (int q = 0; q < HID; ++q) acc += W_gat[o * HID + q] * (b_enc[q] + b_z[q]);
    bvec[o] = acc;
}

// K3: init aggregation output with b_gat; init segment max/sum.
__global__ void init_buffers(float* __restrict__ outagg,
                             const float* __restrict__ b_gat,
                             float* __restrict__ mbuf,
                             float* __restrict__ ssum, int N) {
    int idx = blockIdx.x * blockDim.x + threadIdx.x;
    if (idx < N * HID) outagg[idx] = b_gat[idx & (HID - 1)];
    if (idx < N * HEADS) { mbuf[idx] = -__builtin_inff(); ssum[idx] = 0.0f; }
}

// ---------------------------------------------------------------------------
// K1: xh[N,256] = [x | z[batch]] @ W_comb.T + bvec   (fused encoder+GAT linear)
// One 16-node tile per block (8 waves). A panel (f32) staged in LDS with
// async global->LDS b128 copies; each wave owns two 16-col tiles; 6 WMMA
// K-steps (K=192), f32->f16 conversion during fragment build.
// ---------------------------------------------------------------------------
__global__ void fused_encode_gemm(const float* __restrict__ x,
                                  const int* __restrict__ batch,
                                  const float* __restrict__ z,
                                  const _Float16* __restrict__ wsw,
                                  const float* __restrict__ bvec,
                                  float* __restrict__ xh, int N) {
    __shared__ float Ap[16 * KCOMB];              // 12 KB f32 panel
    const int tid  = threadIdx.x;
    const int base = blockIdx.x * 16;

    // Stage A panel: 16 rows x 48 float4 chunks = 768 = 3 x 256 (exact)
    #pragma unroll
    for (int it = 0; it < 3; ++it) {
        int c = it * 256 + tid;
        int m = c / 48, pos = c - m * 48;         // pos in float4 units
        int node = base + m; if (node >= N) node = N - 1;
        const void* g = (pos < 32)
            ? (const void*)(x + node * NODE_DIM + pos * 4)
            : (const void*)(z + batch[node] * LAT + (pos - 32) * 4);
        async_copy_b128(&Ap[m * KCOMB + pos * 4], g);
    }
    wait_async0();
    __syncthreads();

    const int wave = tid >> 5, lane = tid & 31;
    const int am  = lane & 15;
    const int klo = (lane < 16) ? 0 : 8;          // A chunk base within K-step

    for (int tt = 0; tt < 2; ++tt) {
        const int t = wave * 2 + tt;              // N-tile 0..15
        const int col = t * 16 + (lane & 15);
        const float bias = bvec[col];
        v8f acc;
        #pragma unroll
        for (int r = 0; r < 8; ++r) acc[r] = bias;

        #pragma unroll
        for (int s = 0; s < 6; ++s) {
            const v16h a = make_afrag(Ap, KCOMB, am, s, klo);
            const v16h b = ((const v16h*)wsw)[(t * 6 + s) * 32 + lane];
            acc = __builtin_amdgcn_wmma_f32_16x16x32_f16(
                      false, a, false, b, (short)0, acc, false, false);
        }
        // scatter D: elem r -> node base+M, col = t*16 + (lane&15)
        const int row0 = base + ((lane < 16) ? 0 : 8);
        if (base + 16 <= N) {                     // full tile: unguarded stores
            #pragma unroll
            for (int r = 0; r < 8; ++r) xh[(row0 + r) * HID + col] = acc[r];
        } else {
            #pragma unroll
            for (int r = 0; r < 8; ++r)
                if (row0 + r < N) xh[(row0 + r) * HID + col] = acc[r];
        }
    }
}

// K2: per-node, per-head attention coefficients
__global__ void att_coef(const float* __restrict__ xh,
                         const float* __restrict__ att_src,
                         const float* __restrict__ att_dst,
                         float* __restrict__ a_src,
                         float* __restrict__ a_dst, int N) {
    int idx = blockIdx.x * blockDim.x + threadIdx.x;
    if (idx >= N * HEADS) return;
    int n = idx >> 2, h = idx & 3;
    const float* v = xh + n * HID + h * CPH;
    float s0 = 0.0f, s1 = 0.0f;
    for (int c = 0; c < CPH; ++c) {
        float xv = v[c];
        s0 += xv * att_src[h * CPH + c];
        s1 += xv * att_dst[h * CPH + c];
    }
    a_src[idx] = s0; a_dst[idx] = s1;
}

// K4: segment max over incoming edges (incl. self loops)
__global__ void edge_max(const int* __restrict__ ei, int E, int N,
                         const float* __restrict__ a_src,
                         const float* __restrict__ a_dst,
                         float* __restrict__ mbuf) {
    int i = blockIdx.x * blockDim.x + threadIdx.x;
    if (i >= E + N) return;
    int s = (i < E) ? ei[i] : (i - E);
    int d = (i < E) ? ei[E + i] : (i - E);
    #pragma unroll
    for (int h = 0; h < HEADS; ++h) {
        float e = a_src[s * HEADS + h] + a_dst[d * HEADS + h];
        e = (e > 0.0f) ? e : NEG_SLOPE * e;
        atomicMaxF(&mbuf[d * HEADS + h], e);
    }
}

// K5: exp(e - max) per edge, accumulate segment sums
__global__ void edge_exp(const int* __restrict__ ei, int E, int N,
                         const float* __restrict__ a_src,
                         const float* __restrict__ a_dst,
                         const float* __restrict__ mbuf,
                         float* __restrict__ exbuf,
                         float* __restrict__ ssum) {
    int i = blockIdx.x * blockDim.x + threadIdx.x;
    if (i >= E + N) return;
    int s = (i < E) ? ei[i] : (i - E);
    int d = (i < E) ? ei[E + i] : (i - E);
    #pragma unroll
    for (int h = 0; h < HEADS; ++h) {
        float e = a_src[s * HEADS + h] + a_dst[d * HEADS + h];
        e = (e > 0.0f) ? e : NEG_SLOPE * e;
        float ex = __expf(e - mbuf[d * HEADS + h]);
        exbuf[i * HEADS + h] = ex;
        atomicAdd(&ssum[d * HEADS + h], ex);
    }
}

// K6: wave-per-edge weighted scatter-add. Edge index is wave-uniform ->
// force scalar so src/dst fetches become s_loads. Lane handles 8 contiguous
// channels (one head per 8-lane group): float4 gathers + 8 f32 atomic adds.
__global__ void edge_aggregate(const int* __restrict__ ei, int E, int N,
                               const float* __restrict__ xh,
                               const float* __restrict__ exbuf,
                               const float* __restrict__ ssum,
                               float* __restrict__ outagg) {
    int i = blockIdx.x * 8 + (threadIdx.x >> 5);
    i = __builtin_amdgcn_readfirstlane(i);        // wave-uniform
    if (i >= E + N) return;
    int lane = threadIdx.x & 31;
    int s = (i < E) ? ei[i] : (i - E);
    int d = (i < E) ? ei[E + i] : (i - E);
    int h = lane >> 3;
    float alpha = exbuf[i * HEADS + h] / ssum[d * HEADS + h];
    int c0 = lane * 8;
    const float4 v0 = *(const float4*)(xh + s * HID + c0);
    const float4 v1 = *(const float4*)(xh + s * HID + c0 + 4);
    float* o = outagg + d * HID + c0;
    atomicAdd(o + 0, v0.x * alpha); atomicAdd(o + 1, v0.y * alpha);
    atomicAdd(o + 2, v0.z * alpha); atomicAdd(o + 3, v0.w * alpha);
    atomicAdd(o + 4, v1.x * alpha); atomicAdd(o + 5, v1.y * alpha);
    atomicAdd(o + 6, v1.z * alpha); atomicAdd(o + 7, v1.w * alpha);
}

// ---------------------------------------------------------------------------
// K7: logits = relu(outagg @ W_e1.T + b_e1) @ W_e2.T + b_e2, fused via WMMA.
// A panel staged with async global->LDS copies; hh stays in registers;
// column dot with W_e2 reduced through LDS ds_add_f32.
// ---------------------------------------------------------------------------
__global__ void final_mlp(const float* __restrict__ outagg,
                          const _Float16* __restrict__ we1sw,
                          const float* __restrict__ b_e1,
                          const float* __restrict__ W_e2,
                          const float* __restrict__ b_e2,
                          float* __restrict__ logits, int N) {
    __shared__ float Ap[16 * HID];                // 16 KB f32 panel
    __shared__ float lsum[16];
    const int tid  = threadIdx.x;
    const int base = blockIdx.x * 16;

    // Stage A panel: 16 rows x 64 float4 chunks = 1024 = 4 x 256 (exact)
    #pragma unroll
    for (int it = 0; it < 4; ++it) {
        int c = it * 256 + tid;
        int m = c >> 6, pos = c & 63;             // pos in float4 units
        int node = base + m; if (node >= N) node = N - 1;
        async_copy_b128(&Ap[m * HID + pos * 4], outagg + node * HID + pos * 4);
    }
    if (tid < 16) lsum[tid] = 0.0f;
    wait_async0();
    __syncthreads();

    const int wave = tid >> 5, lane = tid & 31;
    const int am  = lane & 15;
    const int klo = (lane < 16) ? 0 : 8;

    for (int tt = 0; tt < 2; ++tt) {
        const int t = wave * 2 + tt;
        const int col = t * 16 + (lane & 15);
        const float bias = b_e1[col];
        v8f acc;
        #pragma unroll
        for (int r = 0; r < 8; ++r) acc[r] = bias;

        #pragma unroll
        for (int s = 0; s < 8; ++s) {
            const v16h a = make_afrag(Ap, HID, am, s, klo);
            const v16h b = ((const v16h*)we1sw)[(t * 8 + s) * 32 + lane];
            acc = __builtin_amdgcn_wmma_f32_16x16x32_f16(
                      false, a, false, b, (short)0, acc, false, false);
        }
        const float w2 = W_e2[col];
        #pragma unroll
        for (int r = 0; r < 8; ++r) {
            float hh = acc[r] > 0.0f ? acc[r] : 0.0f;     // relu
            int m = (lane < 16) ? r : r + 8;
            atomicAdd(&lsum[m], hh * w2);                  // ds_add_f32
        }
    }
    __syncthreads();
    if (tid < 16) {
        int node = base + tid;
        if (node < N) logits[node] = lsum[tid] + b_e2[0];
    }
}

// ---------------------------------------------------------------------------
extern "C" void kernel_launch(void* const* d_in, const int* in_sizes, int n_in,
                              void* d_out, int out_size, void* d_ws, size_t ws_size,
                              hipStream_t stream) {
    (void)n_in; (void)out_size; (void)ws_size;
    const float* x       = (const float*)d_in[0];
    const int*   ei      = (const int*)  d_in[1];
    const int*   batch   = (const int*)  d_in[2];
    const float* z       = (const float*)d_in[3];
    const float* W_enc   = (const float*)d_in[4];
    const float* b_enc   = (const float*)d_in[5];
    const float* W_z     = (const float*)d_in[6];
    const float* b_z     = (const float*)d_in[7];
    const float* W_gat   = (const float*)d_in[8];
    const float* att_src = (const float*)d_in[9];
    const float* att_dst = (const float*)d_in[10];
    const float* b_gat   = (const float*)d_in[11];
    const float* W_e1    = (const float*)d_in[12];
    const float* b_e1    = (const float*)d_in[13];
    const float* W_e2    = (const float*)d_in[14];
    const float* b_e2    = (const float*)d_in[15];
    float* logits = (float*)d_out;

    const int N = in_sizes[0] / NODE_DIM;
    const int E = in_sizes[1] / 2;
    const int EN = E + N;

    // Workspace carve-out (256B aligned slabs)
    char* ws = (char*)d_ws;
    size_t off = 0;
    auto carve = [&](size_t bytes) -> char* {
        char* p = ws + off;
        off = (off + bytes + 255) & ~(size_t)255;
        return p;
    };
    float*    xh     = (float*)   carve((size_t)N * HID * 4);
    float*    outagg = (float*)   carve((size_t)N * HID * 4);
    float*    a_src  = (float*)   carve((size_t)N * HEADS * 4);
    float*    a_dst  = (float*)   carve((size_t)N * HEADS * 4);
    float*    mbuf   = (float*)   carve((size_t)N * HEADS * 4);
    float*    ssum   = (float*)   carve((size_t)N * HEADS * 4);
    float*    exbuf  = (float*)   carve((size_t)EN * HEADS * 4);
    _Float16* wcomb  = (_Float16*)carve((size_t)16 * 6 * 32 * 16 * 2);
    _Float16* we1sw  = (_Float16*)carve((size_t)16 * 8 * 32 * 16 * 2);
    float*    bvec   = (float*)   carve(HID * 4);

    // Weight prep (tiny)
    prep_wcomb<<<(16 * 6 * 32 * 16 + 255) / 256, 256, 0, stream>>>(W_gat, W_enc, W_z, wcomb);
    prep_we1  <<<(16 * 8 * 32 * 16 + 255) / 256, 256, 0, stream>>>(W_e1, we1sw);
    prep_bvec <<<1, 256, 0, stream>>>(W_gat, b_enc, b_z, bvec);

    // Init accumulators
    init_buffers<<<(N * HID + 255) / 256, 256, 0, stream>>>(outagg, b_gat, mbuf, ssum, N);

    // Fused encoder + GAT linear (WMMA, async LDS staging)
    fused_encode_gemm<<<(N + 15) / 16, 256, 0, stream>>>(x, batch, z, wcomb, bvec, xh, N);

    // Attention coefficients
    att_coef<<<(N * HEADS + 255) / 256, 256, 0, stream>>>(xh, att_src, att_dst, a_src, a_dst, N);

    // Segment softmax + weighted scatter aggregation
    edge_max<<<(EN + 255) / 256, 256, 0, stream>>>(ei, E, N, a_src, a_dst, mbuf);
    edge_exp<<<(EN + 255) / 256, 256, 0, stream>>>(ei, E, N, a_src, a_dst, mbuf, exbuf, ssum);
    edge_aggregate<<<(EN + 7) / 8, 256, 0, stream>>>(ei, E, N, xh, exbuf, ssum, outagg);

    // Fused edge-predictor MLP (WMMA, async LDS staging) -> logits
    final_mlp<<<(N + 15) / 16, 256, 0, stream>>>(outagg, we1sw, b_e1, W_e2, b_e2, logits, N);
}
